// Traditional_GCN_12292196402058
// MI455X (gfx1250) — matrix operand order repaired and checked by reading, not compiled, not run on hardware
//
#include <hip/hip_runtime.h>
#include <hip/hip_bf16.h>
#include <stdint.h>

#define N_NODES 8192
#define F_IN    128
#define HID     32
#define N_SEL   4096
#define BN_EPS  1e-5f
#define KSLICES 8
#define KSPAN   (N_NODES / KSLICES)      // 1024

typedef __attribute__((ext_vector_type(16))) __bf16 v16bf;
typedef __attribute__((ext_vector_type(8)))  __bf16 v8bf;
typedef __attribute__((ext_vector_type(8)))  float  v8f;

static __device__ __forceinline__ uint16_t f2bf(float f) {
    union { float f; uint32_t u; } v; v.f = f;
    uint32_t u = v.u;
    // round-to-nearest-even fp32 -> bf16
    uint32_t r = (u + 0x7fffu + ((u >> 16) & 1u)) >> 16;
    return (uint16_t)r;
}

// ---------------------------------------------------------------------------
// 1) d_inv_sqrt[i] = rsqrt(sum_j A[i][j] + 1)   (the +1 is the added identity)
// ---------------------------------------------------------------------------
__global__ void k_rowsum(const float* __restrict__ A, float* __restrict__ dis) {
    __shared__ float red[256];
    int i = blockIdx.x;
    const float* row = A + (size_t)i * N_NODES;
    float s = 0.f;
    for (int j = threadIdx.x; j < N_NODES; j += 256) s += row[j];
    red[threadIdx.x] = s;
    __syncthreads();
    for (int off = 128; off > 0; off >>= 1) {
        if (threadIdx.x < off) red[threadIdx.x] += red[threadIdx.x + off];
        __syncthreads();
    }
    if (threadIdx.x == 0) dis[i] = rsqrtf(red[0] + 1.0f);
}

// ---------------------------------------------------------------------------
// 2) adj_bf16[i][j] = bf16( dis[i] * (A[i][j] + (i==j)) * dis[j] )
// ---------------------------------------------------------------------------
__global__ void k_normadj(const float* __restrict__ A, const float* __restrict__ dis,
                          uint16_t* __restrict__ adjb) {
    size_t t = (size_t)blockIdx.x * blockDim.x + threadIdx.x;
    int i  = (int)(t >> 10);            // 1024 groups of 8 per row
    int jb = ((int)t & 1023) << 3;
    const float* p = A + (size_t)i * N_NODES + jb;
    float di = dis[i];
    union { uint16_t u[8]; uint4 v; } o;
#pragma unroll
    for (int e = 0; e < 8; ++e) {
        int j = jb + e;
        float a = p[e] + ((j == i) ? 1.0f : 0.0f);
        o.u[e] = f2bf(di * dis[j] * a);
    }
    *(uint4*)(adjb + (size_t)i * N_NODES + jb) = o.v;
}

// ---------------------------------------------------------------------------
// 3) Bt[n][k] = bf16( (H @ W)[k][n] )   (transposed: WMMA B frags contiguous)
// ---------------------------------------------------------------------------
__global__ void k_xw_bt(const float* __restrict__ H, const float* __restrict__ W,
                        int fin, uint16_t* __restrict__ bt) {
    extern __shared__ float wl[];       // fin * 32 floats
    for (int t = threadIdx.x; t < fin * HID; t += blockDim.x) wl[t] = W[t];
    __syncthreads();
    int n = threadIdx.x & 31;
    int k = blockIdx.x * 8 + (threadIdx.x >> 5);
    const float* hr = H + (size_t)k * fin;
    float acc = 0.f;
    for (int i = 0; i < fin; ++i) acc += hr[i] * wl[i * HID + n];
    bt[(size_t)n * N_NODES + k] = f2bf(acc);
}

// ---------------------------------------------------------------------------
// 4) partAH[slice] = adj_bf16[:, Kslice] @ Bt^T[Kslice, :]
//    via v_wmma_f32_16x16x32_bf16.  Grid (64, 8): x covers 512 M-tiles
//    (8 waves/block), y = K slice.  4096 waves in flight.
// ---------------------------------------------------------------------------
static __device__ __forceinline__ v16bf load_a_frag(const uint16_t* p) {
    // elements 0..7 = K+0..7 chunk, 8..15 = K+8..15 chunk (contiguous 16 K)
    v8bf lo = *(const v8bf*)p;
    v8bf hi = *(const v8bf*)(p + 8);
    v16bf r;
#pragma unroll
    for (int i = 0; i < 8; ++i) { r[i] = lo[i]; r[i + 8] = hi[i]; }
    return r;
}

__global__ void k_agg_wmma(const uint16_t* __restrict__ adjb,
                           const uint16_t* __restrict__ bt,
                           float* __restrict__ partAH) {
    int wave = (int)((blockIdx.x * blockDim.x + threadIdx.x) >> 5); // M tile 0..511
    int lane = threadIdx.x & 31;
    int half = lane >> 4;                 // 0 or 1 (half-wave)
    int l15  = lane & 15;
    int kbase = blockIdx.y * KSPAN;
    float* out = partAH + (size_t)blockIdx.y * (N_NODES * HID);

    const uint16_t* arow = adjb + (size_t)(wave * 16 + l15) * N_NODES;
    int akoff = half << 4;                // A: +0 / +16 (contiguous 16 K)
    int bkoff = half << 4;                // B: +0 / +16 (contiguous 16 K)
    const uint16_t* b0 = bt + (size_t)l15 * N_NODES;
    const uint16_t* b1 = bt + (size_t)(l15 + 16) * N_NODES;

    v8f acc0 = {}, acc1 = {};
#pragma unroll 2
    for (int kt = kbase; kt < kbase + KSPAN; kt += 32) {
        v16bf a  = load_a_frag(arow + kt + akoff);
        v16bf f0 = *(const v16bf*)(b0 + kt + bkoff);
        v16bf f1 = *(const v16bf*)(b1 + kt + bkoff);
        acc0 = __builtin_amdgcn_wmma_f32_16x16x32_bf16(false, a, false, f0,
                                                       (short)0, acc0, false, false);
        acc1 = __builtin_amdgcn_wmma_f32_16x16x32_bf16(false, a, false, f1,
                                                       (short)0, acc1, false, false);
    }
    // C layout: VGPR j -> row = tile*16 + j + 8*half, col = lane&15 (+16 acc1)
    int orow = wave * 16 + (half << 3);
#pragma unroll
    for (int j = 0; j < 8; ++j) {
        out[(size_t)(orow + j) * HID + l15]      = acc0[j];
        out[(size_t)(orow + j) * HID + l15 + 16] = acc1[j];
    }
}

// ---------------------------------------------------------------------------
// 4b) AH = sum over the 8 K-slice partials (float4 vectorized)
// ---------------------------------------------------------------------------
__global__ void k_sumK(const float* __restrict__ partAH, float* __restrict__ AH) {
    int idx = blockIdx.x * blockDim.x + threadIdx.x;     // float4 index
    const float4* p = (const float4*)partAH;
    float4 s = p[idx];
#pragma unroll
    for (int t = 1; t < KSLICES; ++t) {
        float4 v = p[idx + (size_t)t * (N_NODES * HID / 4)];
        s.x += v.x; s.y += v.y; s.z += v.z; s.w += v.w;
    }
    ((float4*)AH)[idx] = s;
}

// ---------------------------------------------------------------------------
// 5) Per-column partial sums / sums of squares (64 blocks, deterministic tree)
// ---------------------------------------------------------------------------
__global__ void k_colstats(const float* __restrict__ X, int nrows,
                           float* __restrict__ part) {
    __shared__ float ssum[8][32];
    __shared__ float ssq[8][32];
    int col = threadIdx.x & 31;
    int rl  = threadIdx.x >> 5;           // 0..7
    int rpb = nrows >> 6;                 // rows per block (64 blocks)
    int r0  = blockIdx.x * rpb;
    float s = 0.f, q = 0.f;
    for (int r = rl; r < rpb; r += 8) {
        float v = X[(size_t)(r0 + r) * HID + col];
        s += v; q += v * v;
    }
    ssum[rl][col] = s; ssq[rl][col] = q;
    __syncthreads();
    if (rl == 0) {
#pragma unroll
        for (int j = 1; j < 8; ++j) { s += ssum[j][col]; q += ssq[j][col]; }
        part[blockIdx.x * 64 + col]      = s;
        part[blockIdx.x * 64 + 32 + col] = q;
    }
}

// ---------------------------------------------------------------------------
// 6) BN (training stats, biased var) + ReLU
// ---------------------------------------------------------------------------
__global__ void k_bn_relu(const float* __restrict__ X, const float* __restrict__ part,
                          const float* __restrict__ g, const float* __restrict__ b,
                          int nrows, float* __restrict__ Y) {
    __shared__ float scale[32], shift[32];
    if (threadIdx.x < 32) {
        float s = 0.f, q = 0.f;
        for (int blk = 0; blk < 64; ++blk) {
            s += part[blk * 64 + threadIdx.x];
            q += part[blk * 64 + 32 + threadIdx.x];
        }
        float inv  = 1.0f / (float)nrows;
        float mean = s * inv;
        float var  = q * inv - mean * mean;
        float sc   = g[threadIdx.x] * rsqrtf(var + BN_EPS);
        scale[threadIdx.x] = sc;
        shift[threadIdx.x] = b[threadIdx.x] - mean * sc;
    }
    __syncthreads();
    int total  = nrows * HID;
    int stride = gridDim.x * blockDim.x;
    for (int idx = blockIdx.x * blockDim.x + threadIdx.x; idx < total; idx += stride) {
        int col = idx & 31;
        float v = X[idx] * scale[col] + shift[col];
        Y[idx] = v > 0.f ? v : 0.f;
    }
}

// ---------------------------------------------------------------------------
// 7) S[r] = H[idx[r]] @ W6^T + bias6    (torch Linear: W is [out][in])
// ---------------------------------------------------------------------------
__global__ void k_sel_lin(const float* __restrict__ H, const int* __restrict__ idx,
                          const float* __restrict__ W, const float* __restrict__ bias,
                          float* __restrict__ S) {
    __shared__ float wl[HID * HID];
    __shared__ float bl[HID];
    for (int t = threadIdx.x; t < HID * HID; t += blockDim.x) wl[t] = W[t];
    if (threadIdx.x < HID) bl[threadIdx.x] = bias[threadIdx.x];
    __syncthreads();
    int o = threadIdx.x & 31;
    int r = blockIdx.x * 8 + (threadIdx.x >> 5);
    const float* hr = H + (size_t)idx[r] * HID;
    float acc = bl[o];
#pragma unroll
    for (int i = 0; i < HID; ++i) acc += hr[i] * wl[o * HID + i];
    S[(size_t)r * HID + o] = acc;
}

// ---------------------------------------------------------------------------
// 8) out[r] = softmax( S2[r] @ W7^T + bias7 )
// ---------------------------------------------------------------------------
__global__ void k_lin_softmax(const float* __restrict__ S, const float* __restrict__ W,
                              const float* __restrict__ bias, float* __restrict__ out) {
    __shared__ float wl[HID * HID];
    __shared__ float bl[HID];
    for (int t = threadIdx.x; t < HID * HID; t += blockDim.x) wl[t] = W[t];
    if (threadIdx.x < HID) bl[threadIdx.x] = bias[threadIdx.x];
    __syncthreads();
    int r = blockIdx.x * blockDim.x + threadIdx.x;
    float h[HID];
#pragma unroll
    for (int i = 0; i < HID; ++i) h[i] = S[(size_t)r * HID + i];
    float z[HID];
    float mx = -3.4e38f;
#pragma unroll
    for (int o = 0; o < HID; ++o) {
        float acc = bl[o];
#pragma unroll
        for (int i = 0; i < HID; ++i) acc += h[i] * wl[o * HID + i];
        z[o] = acc;
        mx = fmaxf(mx, acc);
    }
    float sum = 0.f;
#pragma unroll
    for (int o = 0; o < HID; ++o) { z[o] = __expf(z[o] - mx); sum += z[o]; }
    float inv = 1.0f / sum;
#pragma unroll
    for (int o = 0; o < HID; ++o) out[(size_t)r * HID + o] = z[o] * inv;
}

// ---------------------------------------------------------------------------
extern "C" void kernel_launch(void* const* d_in, const int* in_sizes, int n_in,
                              void* d_out, int out_size, void* d_ws, size_t ws_size,
                              hipStream_t stream) {
    (void)in_sizes; (void)n_in; (void)out_size; (void)ws_size;

    const float* x   = (const float*)d_in[0];
    const float* adj = (const float*)d_in[1];
    const int*   sel = (const int*)d_in[2];
    // params flattened in dict insertion order: (W,g,b) x5, W6, bias6, g6, b6, W7, bias7
    const float* Wl[5]; const float* gl[5]; const float* bl[5];
    for (int l = 0; l < 5; ++l) {
        Wl[l] = (const float*)d_in[3 + 3 * l];
        gl[l] = (const float*)d_in[4 + 3 * l];
        bl[l] = (const float*)d_in[5 + 3 * l];
    }
    const float* W6    = (const float*)d_in[18];
    const float* bias6 = (const float*)d_in[19];
    const float* g6    = (const float*)d_in[20];
    const float* b6    = (const float*)d_in[21];
    const float* W7    = (const float*)d_in[22];
    const float* bias7 = (const float*)d_in[23];

    // workspace layout
    char* base = (char*)d_ws;
    uint16_t* adjb   = (uint16_t*)(base);                            // 128 MB
    uint16_t* bt     = (uint16_t*)(base + (size_t)134217728);        // 512 KB
    float*    partAH = (float*)   (base + (size_t)134742016);        // 8 MB
    float*    AH     = (float*)   (base + (size_t)143130624);        // 1 MB
    float*    Hb     = (float*)   (base + (size_t)144179200);        // 1 MB
    float*    part   = (float*)   (base + (size_t)145227776);        // 16 KB
    float*    S1     = (float*)   (base + (size_t)145244160);        // 512 KB
    float*    S2     = (float*)   (base + (size_t)145768448);        // 512 KB
    float*    dis    = (float*)   (base + (size_t)146292736);        // 32 KB

    // --- normalized adjacency, converted once to bf16 (L2-resident after) ---
    k_rowsum <<<N_NODES, 256, 0, stream>>>(adj, dis);
    k_normadj<<<(N_NODES * (N_NODES / 8)) / 256, 256, 0, stream>>>(adj, dis, adjb);

    // --- 5 GCN layers ---
    const float* Hin = x;
    int fin = F_IN;
    for (int l = 0; l < 5; ++l) {
        k_xw_bt   <<<N_NODES / 8, 256, fin * HID * sizeof(float), stream>>>(Hin, Wl[l], fin, bt);
        k_agg_wmma<<<dim3(64, KSLICES), 256, 0, stream>>>(adjb, bt, partAH);
        k_sumK    <<<(N_NODES * HID / 4) / 256, 256, 0, stream>>>(partAH, AH);
        k_colstats<<<64, 256, 0, stream>>>(AH, N_NODES, part);
        k_bn_relu <<<1024, 256, 0, stream>>>(AH, part, gl[l], bl[l], N_NODES, Hb);
        Hin = Hb; fin = HID;
    }

    // --- tail: gather + linear6 + BN + ReLU + linear7 + softmax ---
    k_sel_lin    <<<N_SEL / 8, 256, 0, stream>>>(Hb, sel, W6, bias6, S1);
    k_colstats   <<<64, 256, 0, stream>>>(S1, N_SEL, part);
    k_bn_relu    <<<512, 256, 0, stream>>>(S1, part, g6, b6, N_SEL, S2);
    k_lin_softmax<<<N_SEL / 256, 256, 0, stream>>>(S2, W7, bias7, (float*)d_out);
}
